// GAT_81801947119753
// MI455X (gfx1250) — matrix operand order repaired and checked by reading, not compiled, not run on hardware
//
#include <hip/hip_runtime.h>
#include <hip/hip_bf16.h>
#include <math.h>

typedef __attribute__((ext_vector_type(16))) _Float16 v16h;
typedef __attribute__((ext_vector_type(8)))  _Float16 v8h;
typedef __attribute__((ext_vector_type(8)))  float    v8f;
typedef __attribute__((ext_vector_type(4)))  float    v4f;

static __host__ __device__ inline int ceil_div(int a, int b) { return (a + b - 1) / b; }

// ---------------------------------------------------------------------------
// fp32 -> fp16 conversion (grid-stride)
// ---------------------------------------------------------------------------
__global__ void gat_cvt_f32_f16(const float* __restrict__ in, _Float16* __restrict__ out, int n) {
    int i = blockIdx.x * blockDim.x + threadIdx.x;
    int stride = gridDim.x * blockDim.x;
    for (; i < n; i += stride) out[i] = (_Float16)in[i];
}

// ---------------------------------------------------------------------------
// Pack W [K, M] fp32 (row-major) into WMMA B-fragment order, fp16:
//   Bp[((k/16)*(M/16) + col/16)*256 + (col%16)*16 + (k%16)]
// so each lane's 16 B elements (one k-chunk of its column) are contiguous.
// ---------------------------------------------------------------------------
__global__ void gat_pack_B(const float* __restrict__ W, _Float16* __restrict__ Bp,
                           int K, int M) {
    int i = blockIdx.x * blockDim.x + threadIdx.x;
    if (i >= K * M) return;
    int k = i / M, col = i % M;
    int idx = ((k >> 4) * (M >> 4) + (col >> 4)) * 256 + (col & 15) * 16 + (k & 15);
    Bp[idx] = (_Float16)W[i];
}

// ---------------------------------------------------------------------------
// zero fp32 buffer, float4 granularity (count n4 = #float4)
// ---------------------------------------------------------------------------
__global__ void gat_zero_f32(float* __restrict__ p, int n4) {
    int i = blockIdx.x * blockDim.x + threadIdx.x;
    int stride = gridDim.x * blockDim.x;
    v4f z = {0.f, 0.f, 0.f, 0.f};
    for (; i < n4; i += stride) ((v4f*)p)[i] = z;
}

// ---------------------------------------------------------------------------
// GEMM: D[N,M] (fp16) = A[N,K=256] (fp16) @ B (packed fragments), fp32 accum.
// One wave computes a 16x32 output strip (two 16x16 tiles sharing the A
// fragment); 4 waves (128 threads) per block.  K loop fully unrolled ->
// 16 static v_wmma_f32_16x16x32_f16 per kernel.
// ---------------------------------------------------------------------------
__global__ void gat_gemm_wmma(const _Float16* __restrict__ A,
                              const _Float16* __restrict__ Bp,
                              _Float16* __restrict__ D,
                              int N, int M) {
    const int K = 256;
    const int lane   = threadIdx.x & 31;
    const int wave   = threadIdx.x >> 5;
    const int tile_m = blockIdx.x * 4 + wave;
    const int tile_n0 = blockIdx.y * 2;          // two adjacent N-tiles per wave
    if (tile_m * 16 >= N) return;                // wave-uniform: EXEC all-1s for WMMA

    const int hs  = lane >> 4;                   // half-select
    const int ml  = lane & 15;
    const int rowA = tile_m * 16 + ml;
    const int ntiles = M >> 4;

    const _Float16* arow = A + (size_t)rowA * K;

    v8f acc0 = {};
    v8f acc1 = {};
#pragma unroll
    for (int k0 = 0; k0 < K; k0 += 32) {
        // A fragment: elems 0..15 = A[row, k0+hs*16 .. +15]
        const v8h a_lo = *(const v8h*)(arow + k0 + hs * 16);
        const v8h a_hi = *(const v8h*)(arow + k0 + hs * 16 + 8);
        v16h a;
#pragma unroll
        for (int j = 0; j < 8; ++j) { a[j] = a_lo[j]; a[8 + j] = a_hi[j]; }

        // B fragments: contiguous 32B per lane from packed layout
        const size_t kblk = (size_t)((k0 >> 4) + hs) * ntiles;
        const v16h b0 = *(const v16h*)(Bp + (kblk + tile_n0) * 256 + ml * 16);
        const v16h b1 = *(const v16h*)(Bp + (kblk + tile_n0 + 1) * 256 + ml * 16);

        acc0 = __builtin_amdgcn_wmma_f32_16x16x32_f16(false, a, false, b0,
                                                      (short)0, acc0, false, false);
        acc1 = __builtin_amdgcn_wmma_f32_16x16x32_f16(false, a, false, b1,
                                                      (short)0, acc1, false, false);
    }

    // D layout: lane group hs, VGPR r -> row = tile_m*16 + hs*8 + r
#pragma unroll
    for (int r = 0; r < 8; ++r) {
        int row = tile_m * 16 + hs * 8 + r;
        D[(size_t)row * M + tile_n0 * 16 + ml]      = (_Float16)acc0[r];
        D[(size_t)row * M + (tile_n0 + 1) * 16 + ml] = (_Float16)acc1[r];
    }
}

// ---------------------------------------------------------------------------
// Per-node attention logits: aS[n,h] = <h[n,h,:], a_src[h,:]>, same for aD.
// Also initializes per-node segment max (-inf) and sum (0).
// ---------------------------------------------------------------------------
__global__ void gat_alpha(const _Float16* __restrict__ H,   // [N, 4*C] fp16
                          const float* __restrict__ a_src,  // [4, C]
                          const float* __restrict__ a_dst,  // [4, C]
                          float* __restrict__ aS, float* __restrict__ aD,
                          float* __restrict__ Mx, float* __restrict__ Sx,
                          int N, int C) {
    int idx = blockIdx.x * blockDim.x + threadIdx.x;
    if (idx >= N * 4) return;
    int n = idx >> 2, h = idx & 3;
    const _Float16* hp = H + (size_t)n * (4 * C) + h * C;
    const float* as = a_src + h * C;
    const float* ad = a_dst + h * C;
    float s1 = 0.f, s2 = 0.f;
    for (int c = 0; c < C; ++c) {
        float v = (float)hp[c];
        s1 += v * as[c];
        s2 += v * ad[c];
    }
    aS[idx] = s1;
    aD[idx] = s2;
    Mx[idx] = -INFINITY;
    Sx[idx] = 0.f;
}

__device__ inline float gat_lrelu(float x) { return x > 0.f ? x : 0.2f * x; }

__device__ inline void gat_atomic_max_f32(float* addr, float val) {
    unsigned int* ua = (unsigned int*)addr;
    unsigned int cur = *ua;
    while (__uint_as_float(cur) < val) {
        unsigned int prev = atomicCAS(ua, cur, __float_as_uint(val));
        if (prev == cur) break;
        cur = prev;
    }
}

// ---------------------------------------------------------------------------
// Edge pass 1: segment max of leaky-relu scores.  One thread per (edge, head).
// ---------------------------------------------------------------------------
__global__ void gat_edge_max(const int* __restrict__ ei,
                             const float* __restrict__ aS, const float* __restrict__ aD,
                             float* __restrict__ Mx, int E) {
    int idx = blockIdx.x * blockDim.x + threadIdx.x;
    if (idx >= E * 4) return;
    int e = idx >> 2, h = idx & 3;
    int src = ei[e], dst = ei[E + e];
    float v = gat_lrelu(aS[src * 4 + h] + aD[dst * 4 + h]);
    gat_atomic_max_f32(&Mx[dst * 4 + h], v);
}

// ---------------------------------------------------------------------------
// Edge pass 2: segment sum of exp(score - max).
// ---------------------------------------------------------------------------
__global__ void gat_edge_sum(const int* __restrict__ ei,
                             const float* __restrict__ aS, const float* __restrict__ aD,
                             const float* __restrict__ Mx, float* __restrict__ Sx, int E) {
    int idx = blockIdx.x * blockDim.x + threadIdx.x;
    if (idx >= E * 4) return;
    int e = idx >> 2, h = idx & 3;
    int src = ei[e], dst = ei[E + e];
    float v = gat_lrelu(aS[src * 4 + h] + aD[dst * 4 + h]);
    atomicAdd(&Sx[dst * 4 + h], expf(v - Mx[dst * 4 + h]));
}

// ---------------------------------------------------------------------------
// Edge pass 3: scatter messages.  One wave per edge; each lane owns MOUT/32
// contiguous channels (always within one head).  h rows (25.6 MB) and ACC
// (<=102 MB) both fit in the 192 MB L2 -> gathers and atomics resolve at L2.
// ---------------------------------------------------------------------------
template <int MOUT>
__global__ void gat_edge_msg(const int* __restrict__ ei,
                             const _Float16* __restrict__ H,
                             const float* __restrict__ aS, const float* __restrict__ aD,
                             const float* __restrict__ Mx, const float* __restrict__ Sx,
                             float* __restrict__ ACC, int E) {
    int gwave = (blockIdx.x * blockDim.x + threadIdx.x) >> 5;
    int lane  = threadIdx.x & 31;
    if (gwave >= E) return;
    int src = ei[gwave], dst = ei[E + gwave];

    const int CH = MOUT / 32;     // channels per lane (8 or 16)
    const int C  = MOUT / 4;      // channels per head (64 or 128)
    int c0 = lane * CH;
    int hd = c0 / C;              // head owned by this lane

    const _Float16* hp = H + (size_t)src * MOUT + c0;
    __builtin_prefetch(hp, 0, 1); // global_prefetch_b8

    float e = gat_lrelu(aS[src * 4 + hd] + aD[dst * 4 + hd]);
    float w = expf(e - Mx[dst * 4 + hd]) / (Sx[dst * 4 + hd] + 1e-16f);

    float* op = ACC + (size_t)dst * MOUT + c0;
#pragma unroll
    for (int k = 0; k < CH; ++k)
        atomicAdd(op + k, (float)hp[k] * w);
}

// ---------------------------------------------------------------------------
// Layers 1-3 epilogue: x_next = ELU(acc + b) -> fp16 for next GEMM.
// ---------------------------------------------------------------------------
__global__ void gat_finalize_elu(const float* __restrict__ ACC, const float* __restrict__ b,
                                 _Float16* __restrict__ Xh, int N, int M) {
    int i = blockIdx.x * blockDim.x + threadIdx.x;
    if (i >= N * M) return;
    int j = i % M;
    float v = ACC[i] + b[j];
    v = v > 0.f ? v : (expf(v) - 1.f);
    Xh[i] = (_Float16)v;
}

// ---------------------------------------------------------------------------
// Layer 4 epilogue: mean over heads + bias -> fp32 output [N, 128].
// ---------------------------------------------------------------------------
__global__ void gat_finalize_out(const float* __restrict__ ACC, const float* __restrict__ b,
                                 float* __restrict__ out, int N) {
    int i = blockIdx.x * blockDim.x + threadIdx.x;
    if (i >= N * 128) return;
    int n = i / 128, j = i % 128;
    const float* p = ACC + (size_t)n * 512;
    out[i] = 0.25f * (p[j] + p[128 + j] + p[256 + j] + p[384 + j]) + b[j];
}

// ---------------------------------------------------------------------------
// Host orchestration
// ---------------------------------------------------------------------------
extern "C" void kernel_launch(void* const* d_in, const int* in_sizes, int n_in,
                              void* d_out, int out_size, void* d_ws, size_t ws_size,
                              hipStream_t stream) {
    const int N = 50000, E = 800000, K = 256;

    const float* x  = (const float*)d_in[0];
    const int*   ei = (const int*)d_in[1];
    const float* Ws[4]   = {(const float*)d_in[2],  (const float*)d_in[6],
                            (const float*)d_in[10], (const float*)d_in[14]};
    const float* aSrc[4] = {(const float*)d_in[3],  (const float*)d_in[7],
                            (const float*)d_in[11], (const float*)d_in[15]};
    const float* aDst[4] = {(const float*)d_in[4],  (const float*)d_in[8],
                            (const float*)d_in[12], (const float*)d_in[16]};
    const float* bs[4]   = {(const float*)d_in[5],  (const float*)d_in[9],
                            (const float*)d_in[13], (const float*)d_in[17]};

    // workspace carve-out (256B aligned regions)
    char* ws = (char*)d_ws;
    size_t off = 0;
    auto carve = [&](size_t bytes) -> void* {
        void* p = ws + off;
        off += (bytes + 255) & ~(size_t)255;
        return p;
    };
    _Float16* Xh  = (_Float16*)carve((size_t)N * 256 * sizeof(_Float16));   // GEMM input
    _Float16* Bp  = (_Float16*)carve((size_t)256 * 512 * sizeof(_Float16)); // packed fp16 weights
    _Float16* H16 = (_Float16*)carve((size_t)N * 512 * sizeof(_Float16));   // post-GEMM h
    float*    ACC = (float*)carve((size_t)N * 512 * sizeof(float));         // message accumulator
    float*    aS  = (float*)carve((size_t)N * 4 * sizeof(float));
    float*    aD  = (float*)carve((size_t)N * 4 * sizeof(float));
    float*    Mx  = (float*)carve((size_t)N * 4 * sizeof(float));
    float*    Sx  = (float*)carve((size_t)N * 4 * sizeof(float));
    (void)ws_size; (void)n_in; (void)in_sizes; (void)out_size;

    // input features -> fp16 once
    {
        int n = N * 256;
        gat_cvt_f32_f16<<<dim3(ceil_div(n, 1024)), dim3(256), 0, stream>>>(x, Xh, n);
    }

    for (int l = 0; l < 4; ++l) {
        const int Mout = (l == 3) ? 512 : 256;
        const int C    = Mout / 4;

        // weights -> packed fp16 fragments
        {
            int n = K * Mout;
            gat_pack_B<<<dim3(ceil_div(n, 256)), dim3(256), 0, stream>>>(Ws[l], Bp, K, Mout);
        }

        // h = x @ W  (WMMA f16 -> f32 -> stored f16); 16x32 strip per wave
        {
            dim3 grid(ceil_div(N / 16, 4), Mout / 32);
            gat_gemm_wmma<<<grid, dim3(128), 0, stream>>>(Xh, Bp, H16, N, Mout);
        }

        // zero accumulator
        {
            int n4 = (N * Mout) / 4;
            gat_zero_f32<<<dim3(ceil_div(n4, 1024)), dim3(256), 0, stream>>>(ACC, n4);
        }

        // per-node logits + init max/sum
        gat_alpha<<<dim3(ceil_div(N * 4, 256)), dim3(256), 0, stream>>>(
            H16, aSrc[l], aDst[l], aS, aD, Mx, Sx, N, C);

        // segment softmax over edges
        gat_edge_max<<<dim3(ceil_div(E * 4, 256)), dim3(256), 0, stream>>>(ei, aS, aD, Mx, E);
        gat_edge_sum<<<dim3(ceil_div(E * 4, 256)), dim3(256), 0, stream>>>(ei, aS, aD, Mx, Sx, E);

        // message scatter (one wave per edge)
        {
            dim3 grid(ceil_div(E, 8));   // 8 waves per 256-thread block
            if (Mout == 256)
                gat_edge_msg<256><<<grid, dim3(256), 0, stream>>>(ei, H16, aS, aD, Mx, Sx, ACC, E);
            else
                gat_edge_msg<512><<<grid, dim3(256), 0, stream>>>(ei, H16, aS, aD, Mx, Sx, ACC, E);
        }

        // epilogue
        if (l < 3) {
            int n = N * 256;
            gat_finalize_elu<<<dim3(ceil_div(n, 256)), dim3(256), 0, stream>>>(ACC, bs[l], Xh, N, 256);
        } else {
            int n = N * 128;
            gat_finalize_out<<<dim3(ceil_div(n, 256)), dim3(256), 0, stream>>>(ACC, bs[3], (float*)d_out, N);
        }
    }
}
